// SelfAttention_5798205849867
// MI455X (gfx1250) — matrix operand order repaired and checked by reading, not compiled, not run on hardware
//
#include <hip/hip_runtime.h>

#define DEVI __device__ __forceinline__

typedef __bf16 bf16;
typedef __attribute__((ext_vector_type(16))) __bf16 v16bf;
typedef __attribute__((ext_vector_type(8)))  __bf16 v8bf;
typedef __attribute__((ext_vector_type(8)))  float  v8f;

// async-to-LDS builtin takes int4* in addrspace(1)/(3)
typedef int v4i __attribute__((vector_size(16)));
typedef __attribute__((address_space(1))) v4i* as1_v4i_p;
typedef __attribute__((address_space(3))) v4i* as3_v4i_p;

#if defined(__has_builtin)
#if __has_builtin(__builtin_amdgcn_global_load_async_to_lds_b128) && \
    __has_builtin(__builtin_amdgcn_s_wait_asynccnt)
#define USE_ASYNC_LDS 1
#endif
#endif

// Problem constants (from reference): B=16, C=256, H=W=64 -> HW=4096, G=32
constexpr int NB  = 16;
constexpr int NC  = 256;
constexpr int NHW = 4096;
constexpr int NG  = 32;
constexpr int CPG = NC / NG;          // 8 channels per group
constexpr int BC  = 32;               // key tile for flash attention

DEVI bf16 f2bf(float f) {
    unsigned u = __builtin_bit_cast(unsigned, f);
    unsigned r = u + 0x7FFFu + ((u >> 16) & 1u);   // round-to-nearest-even
    unsigned short h = (unsigned short)(r >> 16);
    return __builtin_bit_cast(bf16, h);
}

// ---- WMMA fragment loaders (CDNA5 wave32 VGPR layouts, CDNA5 ISA §7.12.2) ----
// A-matrix 16x32 bf16, row-major memory with leading dim `ld` (elements).
// lanes 0-15: M=lane, K chunks {0..7, 16..23}; lanes 16-31: M=lane-16, {8..15, 24..31}
DEVI v16bf loadA(const bf16* tile, int ld, int lane) {
    const bf16* p = tile + (size_t)(lane & 15) * ld + ((lane >> 4) << 3);
    union { v16bf v; v8bf h[2]; } f;
    f.h[0] = *(const v8bf*)(p);        // 16B -> b128
    f.h[1] = *(const v8bf*)(p + 16);   // 16B -> b128
    return f.v;
}
// B-matrix 32x16 bf16 (KxN). Memory is "N-major": row n holds 16+ contiguous K.
// lanes 0-15: N=lane, K chunks {0..7, 16..23}; lanes 16-31: N=lane-16, {8..15, 24..31}
DEVI v16bf loadB(const bf16* tile, int ld, int lane) {
    const bf16* p = tile + (size_t)(lane & 15) * ld + ((lane >> 4) << 3);
    union { v16bf v; v8bf h[2]; } f;
    f.h[0] = *(const v8bf*)(p);
    f.h[1] = *(const v8bf*)(p + 16);
    return f.v;
}

DEVI v8f wmma_bf16(v16bf a, v16bf b, v8f c) {
    return __builtin_amdgcn_wmma_f32_16x16x32_bf16(false, a, false, b,
                                                   (short)0, c, false, false);
}

// ---------------- Kernel 1: GroupNorm statistics ----------------
__global__ void gn_stats(const float* __restrict__ x, float* __restrict__ stats) {
    int bg = blockIdx.x;                               // b*NG + g, groups contiguous in C
    const float* p = x + (size_t)bg * (CPG * NHW);     // 32768 floats
    float s = 0.f, ss = 0.f;
    for (int i = threadIdx.x; i < CPG * NHW; i += 256) {
        float v = p[i];
        s += v; ss += v * v;
    }
    for (int m = 1; m < 32; m <<= 1) {
        s  += __shfl_xor(s,  m, 32);
        ss += __shfl_xor(ss, m, 32);
    }
    __shared__ float aS[8], aSS[8];
    int w = threadIdx.x >> 5, l = threadIdx.x & 31;
    if (l == 0) { aS[w] = s; aSS[w] = ss; }
    __syncthreads();
    if (threadIdx.x == 0) {
        float ts = 0.f, tss = 0.f;
        for (int i = 0; i < 8; i++) { ts += aS[i]; tss += aSS[i]; }
        float inv_n = 1.0f / (float)(CPG * NHW);
        float mean = ts * inv_n;
        float var  = tss * inv_n - mean * mean;
        stats[bg * 2 + 0] = mean;
        stats[bg * 2 + 1] = rsqrtf(var + 1e-5f);
    }
}

// ---------------- Kernel 2: weights -> bf16 ----------------
__global__ void w2bf(const float* __restrict__ a, const float* __restrict__ b,
                     const float* __restrict__ c, const float* __restrict__ d,
                     bf16* __restrict__ out) {
    int i = blockIdx.x * 256 + threadIdx.x;            // 0 .. 4*65536-1
    const float* srcs[4] = {a, b, c, d};
    out[i] = f2bf(srcs[i >> 16][i & 65535]);
}

// ---------------- Kernel 3: apply GN + transpose to h[b][hw][c] bf16 ----------------
__global__ void gn_apply(const float* __restrict__ x, const float* __restrict__ stats,
                         const float* __restrict__ gamma, const float* __restrict__ beta,
                         bf16* __restrict__ h) {
    __shared__ float tile[32][33];
    int b = blockIdx.z, c0 = blockIdx.y * 32, hw0 = blockIdx.x * 32;
    int tx = threadIdx.x, ty = threadIdx.y;            // block (32,8)
    #pragma unroll
    for (int i = 0; i < 4; i++) {
        int c = c0 + ty + i * 8;
        int g = c >> 3;
        float mean = stats[(b * NG + g) * 2 + 0];
        float rstd = stats[(b * NG + g) * 2 + 1];
        float v = x[((size_t)b * NC + c) * NHW + hw0 + tx];
        tile[ty + i * 8][tx] = (v - mean) * rstd * gamma[c] + beta[c];
    }
    __syncthreads();
    #pragma unroll
    for (int i = 0; i < 4; i++) {
        int hw = hw0 + ty + i * 8;
        h[((size_t)b * NHW + hw) * NC + c0 + tx] = f2bf(tile[tx][ty + i * 8]);
    }
}

// ---------------- Kernel 4: Q/K/V projections (y = h @ W^T) ----------------
// grid.z selects weight; V is written transposed: Vt[b][c][tok]
__global__ void qkv_gemm(const bf16* __restrict__ h, const bf16* __restrict__ Wbf,
                         bf16* __restrict__ Q, bf16* __restrict__ K,
                         bf16* __restrict__ Vt) {
    int lane = threadIdx.x & 31, wave = threadIdx.x >> 5;
    int which = blockIdx.z;
    const bf16* W = Wbf + (size_t)which * (NC * NC);
    int m0 = blockIdx.x * 128 + wave * 16;             // token-row tile (never crosses batch)
    int n0 = blockIdx.y * 64;                          // output-channel tile
    v8f acc[4] = {};
    for (int k0 = 0; k0 < NC; k0 += 32) {
        v16bf a = loadA(h + (size_t)m0 * NC + k0, NC, lane);
        v16bf bb[4];
        #pragma unroll
        for (int t = 0; t < 4; t++)
            bb[t] = loadB(W + (size_t)(n0 + t * 16) * NC + k0, NC, lane);
        #pragma unroll
        for (int t = 0; t < 4; t++)
            acc[t] = wmma_bf16(a, bb[t], acc[t]);
    }
    int b = m0 >> 12, q = m0 & (NHW - 1);
    int half = lane >> 4, n = lane & 15;
    #pragma unroll
    for (int t = 0; t < 4; t++) {
        #pragma unroll
        for (int j = 0; j < 8; j++) {
            int row = half * 8 + j;                    // M (token)
            int col = n0 + t * 16 + n;                 // N (channel)
            bf16 v = f2bf(acc[t][j]);
            if (which == 0)      Q[((size_t)(m0 + row)) * NC + col] = v;
            else if (which == 1) K[((size_t)(m0 + row)) * NC + col] = v;
            else                 Vt[((size_t)b * NC + col) * NHW + q + row] = v;
        }
    }
}

// ---------------- Kernel 5: flash attention ----------------
// block = 8 waves; each wave: 16 query rows x all 256 channels, online softmax.
__global__ __launch_bounds__(256, 1)
void attn(const bf16* __restrict__ Q, const bf16* __restrict__ Kb,
          const bf16* __restrict__ Vt, bf16* __restrict__ O) {
    __shared__ bf16 sK[BC][NC];        // 16 KB: K tile, [key][channel]
    __shared__ bf16 sV[NC][BC];        // 16 KB: V^T tile, [channel][key]
    __shared__ bf16 sP[8][16][BC];     //  8 KB: per-wave P staging (C-layout -> A-layout)

    int lane = threadIdx.x & 31, wave = threadIdx.x >> 5;
    int half = lane >> 4, n = lane & 15;
    int b = blockIdx.y;
    int q0 = blockIdx.x * 128 + wave * 16;

    const bf16* Qb = Q + ((size_t)b * NHW + q0) * NC;
    v16bf qf[8];
    #pragma unroll
    for (int kk = 0; kk < 8; kk++) qf[kk] = loadA(Qb + kk * 32, NC, lane);

    v8f o[16] = {};
    float mrow[8], lrow[8];
    #pragma unroll
    for (int j = 0; j < 8; j++) { mrow[j] = -1e30f; lrow[j] = 0.f; }

    for (int kt = 0; kt < NHW; kt += BC) {
        // ---- stage K tile (contiguous 16KB) and V^T tile (256 x 64B rows) into LDS ----
        const bf16* ksrc = Kb + ((size_t)b * NHW + kt) * NC;
        const bf16* vsrc = Vt + ((size_t)b * NC + threadIdx.x) * NHW + kt;
#ifdef USE_ASYNC_LDS
        {
            const bf16* ks = ksrc + (size_t)threadIdx.x * 8;   // 16B per thread per sweep
            bf16*       kd = &sK[0][0] + (size_t)threadIdx.x * 8;
            #pragma unroll
            for (int i = 0; i < 4; i++)
                __builtin_amdgcn_global_load_async_to_lds_b128(
                    (as1_v4i_p)(ks + i * 2048), (as3_v4i_p)(kd + i * 2048), 0, 0);
            bf16* vd = &sV[threadIdx.x][0];
            #pragma unroll
            for (int i = 0; i < 4; i++)
                __builtin_amdgcn_global_load_async_to_lds_b128(
                    (as1_v4i_p)(vsrc + i * 8), (as3_v4i_p)(vd + i * 8), 0, 0);
            __builtin_amdgcn_s_wait_asynccnt(0);
        }
#else
        {
            const uint4* src = (const uint4*)ksrc;
            uint4* dst = (uint4*)&sK[0][0];
            #pragma unroll
            for (int i = 0; i < 4; i++) dst[threadIdx.x + 256 * i] = src[threadIdx.x + 256 * i];
            const uint4* vs = (const uint4*)vsrc;
            uint4* vd = (uint4*)&sV[threadIdx.x][0];
            vd[0] = vs[0]; vd[1] = vs[1]; vd[2] = vs[2]; vd[3] = vs[3];
        }
#endif
        __syncthreads();

        // prefetch next K tile into GL2 while we compute on this one
        if (kt + BC < NHW)
            __builtin_prefetch(ksrc + (size_t)BC * NC + (size_t)threadIdx.x * 32, 0, 1);

        // S = Q K^T for two 16-key subtiles
        v8f c0 = {}, c1 = {};
        #pragma unroll
        for (int kk = 0; kk < 8; kk++) {
            v16bf b0 = loadB(&sK[0][kk * 32],  NC, lane);
            v16bf b1 = loadB(&sK[16][kk * 32], NC, lane);
            c0 = wmma_bf16(qf[kk], b0, c0);
            c1 = wmma_bf16(qf[kk], b1, c1);
        }

        // online softmax; row M = half*8+j lives in this lane's 16-lane half
        float corr[8];
        #pragma unroll
        for (int j = 0; j < 8; j++) {
            float s0 = c0[j] * 0.0625f, s1 = c1[j] * 0.0625f;   // scale = C^-0.5
            float tm = fmaxf(s0, s1);
            #pragma unroll
            for (int m = 1; m < 16; m <<= 1) tm = fmaxf(tm, __shfl_xor(tm, m, 32));
            float nm = fmaxf(mrow[j], tm);
            float cr = __expf(mrow[j] - nm);
            float p0 = __expf(s0 - nm), p1 = __expf(s1 - nm);
            float ts = p0 + p1;
            #pragma unroll
            for (int m = 1; m < 16; m <<= 1) ts += __shfl_xor(ts, m, 32);
            lrow[j] = lrow[j] * cr + ts;
            mrow[j] = nm;
            corr[j] = cr;
            sP[wave][half * 8 + j][n]      = f2bf(p0);
            sP[wave][half * 8 + j][n + 16] = f2bf(p1);
        }
        #pragma unroll
        for (int t = 0; t < 16; t++)
            #pragma unroll
            for (int j = 0; j < 8; j++) o[t][j] *= corr[j];

        // O += P @ V  (per-wave P from LDS; V^T tile from LDS)
        v16bf pf = loadA(&sP[wave][0][0], BC, lane);
        #pragma unroll
        for (int t = 0; t < 16; t++) {
            v16bf vf = loadB(&sV[t * 16][0], BC, lane);
            o[t] = wmma_bf16(pf, vf, o[t]);
        }
        __syncthreads();
    }

    float inv[8];
    #pragma unroll
    for (int j = 0; j < 8; j++) inv[j] = 1.0f / lrow[j];
    bf16* Ob = O + ((size_t)b * NHW + q0) * NC;
    #pragma unroll
    for (int t = 0; t < 16; t++)
        #pragma unroll
        for (int j = 0; j < 8; j++)
            Ob[(size_t)(half * 8 + j) * NC + t * 16 + n] = f2bf(o[t][j] * inv[j]);
}

// ---------------- Kernel 6: out = x + O @ Wp^T (transposed back to [b][c][hw]) ----------------
__global__ void proj_out(const bf16* __restrict__ O, const bf16* __restrict__ Wbf,
                         const float* __restrict__ x, float* __restrict__ out) {
    int lane = threadIdx.x & 31, wave = threadIdx.x >> 5;
    int b = blockIdx.z;
    int q0 = blockIdx.x * 128 + wave * 16;
    int o0 = blockIdx.y * 16;
    const bf16* Wp = Wbf + (size_t)3 * (NC * NC);
    const bf16* Ob = O + ((size_t)b * NHW + q0) * NC;
    v8f acc = {};
    for (int k0 = 0; k0 < NC; k0 += 32) {
        v16bf a = loadA(Ob + k0, NC, lane);
        v16bf w = loadB(Wp + (size_t)o0 * NC + k0, NC, lane);
        acc = wmma_bf16(a, w, acc);
    }
    int half = lane >> 4, n = lane & 15;
    int oc = o0 + n;                                   // N = output channel
    size_t base = ((size_t)b * NC + oc) * NHW + q0 + half * 8;  // 8 consecutive hw
    float4 r0 = ((const float4*)(x + base))[0];
    float4 r1 = ((const float4*)(x + base))[1];
    r0.x += acc[0]; r0.y += acc[1]; r0.z += acc[2]; r0.w += acc[3];
    r1.x += acc[4]; r1.y += acc[5]; r1.z += acc[6]; r1.w += acc[7];
    ((float4*)(out + base))[0] = r0;
    ((float4*)(out + base))[1] = r1;
}

extern "C" void kernel_launch(void* const* d_in, const int* in_sizes, int n_in,
                              void* d_out, int out_size, void* d_ws, size_t ws_size,
                              hipStream_t stream) {
    const float* x     = (const float*)d_in[0];
    const float* Wq    = (const float*)d_in[1];
    const float* Wk    = (const float*)d_in[2];
    const float* Wv    = (const float*)d_in[3];
    const float* Wp    = (const float*)d_in[4];
    const float* gamma = (const float*)d_in[5];
    const float* beta  = (const float*)d_in[6];
    float* out = (float*)d_out;

    const size_t NTOK = (size_t)NB * NHW;              // 65536
    char* ws = (char*)d_ws;
    float* stats = (float*)ws;                         // 512*2 floats
    bf16*  Wbf   = (bf16*)(ws + 8192);                 // 4*256*256 bf16 = 512KB
    bf16*  h     = (bf16*)(ws + 8192 + 524288);        // [b][hw][c]
    bf16*  Q     = h  + NTOK * NC;
    bf16*  K     = Q  + NTOK * NC;
    bf16*  Vt    = K  + NTOK * NC;                     // [b][c][hw]
    bf16*  O     = Vt + NTOK * NC;                     // [b][hw][c]

    gn_stats<<<NB * NG, 256, 0, stream>>>(x, stats);
    w2bf<<<(4 * NC * NC) / 256, 256, 0, stream>>>(Wq, Wk, Wv, Wp, Wbf);
    gn_apply<<<dim3(NHW / 32, NC / 32, NB), dim3(32, 8), 0, stream>>>(x, stats, gamma, beta, h);
    qkv_gemm<<<dim3(NTOK / 128, NC / 64, 3), 256, 0, stream>>>(h, Wbf, Q, K, Vt);
    attn<<<dim3(NHW / 128, NB), 256, 0, stream>>>(Q, K, Vt, O);
    proj_out<<<dim3(NHW / 128, NC / 16, NB), 256, 0, stream>>>(O, Wbf, x, out);
}
